// PixelContrastLoss_82188494176949
// MI455X (gfx1250) — matrix operand order repaired and checked by reading, not compiled, not run on hardware
//
#include <hip/hip_runtime.h>
#include <hip/hip_bf16.h>

// ---------------------------------------------------------------------------
// PixelContrastLoss for gfx1250 (MI455X).
//   B=8, P=16384, D=256, C=8, N_VIEW=16  ->  A=64 slots, N=1024 anchors.
//   Pipeline: select -> gather(f32->f16) -> WMMA Gram matrix -> row softmax
//             -> final scalar reduce.
// ---------------------------------------------------------------------------

#define BB 8
#define PP 16384
#define DD 256
#define CC 8
#define NV 16
#define AA (BB * CC)        // 64 slots
#define NN (AA * NV)        // 1024 anchors
#define BP (BB * PP)        // 131072 pixels
#define TEMP_F 0.1f
#define BASE_TEMP_F 0.07f

typedef _Float16 v16h __attribute__((ext_vector_type(16)));
typedef _Float16 v8h  __attribute__((ext_vector_type(8)));
typedef float    v8f  __attribute__((ext_vector_type(8)));

__device__ __forceinline__ unsigned hashu(unsigned x) {
    x ^= x >> 16; x *= 0x7feb352dU;
    x ^= x >> 15; x *= 0x846ca68bU;
    x ^= x >> 16;
    return x;
}

// ---------------------------------------------------------------------------
// Kernel 1: per-slot candidate selection (random top-8 hard + top-8 easy via
// hash scores) and validity (per-image class count > MAX_VIEWS).
// One 256-thread block per slot (64 blocks).
// ---------------------------------------------------------------------------
__global__ void pcl_select(const int* __restrict__ y_hat,
                           const int* __restrict__ y,
                           int* __restrict__ idx_out,     // [AA][NV]
                           int* __restrict__ valid_out) { // [AA]
    const int slot = blockIdx.x;        // 0..63
    const int b = slot >> 3;
    const int c = slot & 7;
    const int tid = threadIdx.x;

    __shared__ int      ired[256];
    __shared__ unsigned rsc[256];
    __shared__ int      rpos[256];
    __shared__ unsigned s_sc[2048];
    __shared__ int      s_id[2048];

    // validity: count of (y_hat[b,:] == c) over image b
    int cnt = 0;
    for (int p = tid; p < PP; p += 256)
        cnt += (y_hat[b * PP + p] == c) ? 1 : 0;
    ired[tid] = cnt;
    __syncthreads();
    for (int s = 128; s > 0; s >>= 1) {
        if (tid < s) ired[tid] += ired[tid + s];
        __syncthreads();
    }
    if (tid == 0) valid_out[slot] = (ired[0] > 100) ? 1 : 0;
    __syncthreads();

    // kind 0 = hard (pred==c, gt!=c), kind 1 = easy (pred==c, gt==c)
    for (int kind = 0; kind < 2; ++kind) {
        unsigned bs[8];
        int      bi[8];
#pragma unroll
        for (int r = 0; r < 8; ++r) { bs[r] = 0u; bi[r] = 0; }

        for (int p = tid; p < BP; p += 256) {
            const int yh = y_hat[p];
            if (yh != c) continue;
            const int yy = y[p];
            const bool cand = (kind == 0) ? (yy != c) : (yy == c);
            if (!cand) continue;
            const unsigned sc =
                hashu((unsigned)p * 0x9E3779B9u ^
                      ((unsigned)(slot * 2 + kind) + 1u) * 0x85EBCA6Bu) | 1u;
            // replace current per-thread minimum of the 8 kept scores
            int mn = 0;
#pragma unroll
            for (int r = 1; r < 8; ++r) if (bs[r] < bs[mn]) mn = r;
            if (sc > bs[mn]) { bs[mn] = sc; bi[mn] = p; }
        }
#pragma unroll
        for (int r = 0; r < 8; ++r) {
            s_sc[tid * 8 + r] = bs[r];
            s_id[tid * 8 + r] = bi[r];
        }
        __syncthreads();

        // 8 rounds of deterministic block-wide argmax over the 2048 survivors
        for (int r = 0; r < 8; ++r) {
            unsigned best = 0u;
            int bpos = -1;
            for (int t = tid; t < 2048; t += 256) {
                const unsigned v = s_sc[t];
                if (v > best || (v == best && bpos >= 0 && t < bpos && v != 0u)) {
                    best = v; bpos = t;
                }
            }
            rsc[tid] = best; rpos[tid] = bpos;
            __syncthreads();
            for (int s = 128; s > 0; s >>= 1) {
                if (tid < s) {
                    const unsigned vo = rsc[tid + s];
                    const int      po = rpos[tid + s];
                    if (vo > rsc[tid] ||
                        (vo == rsc[tid] && po >= 0 &&
                         (rpos[tid] < 0 || po < rpos[tid]))) {
                        rsc[tid] = vo; rpos[tid] = po;
                    }
                }
                __syncthreads();
            }
            if (tid == 0) {
                const int w = rpos[0];
                idx_out[slot * NV + kind * 8 + r] = (w >= 0) ? s_id[w] : 0;
                if (w >= 0) s_sc[w] = 0u;
            }
            __syncthreads();
        }
        __syncthreads();
    }
}

// ---------------------------------------------------------------------------
// Kernel 2: gather selected rows into view-major cf [NN][DD] as f16.
// Row n = v*AA + a  sources  Xf[idx[a][v]].  One block per row.
// ---------------------------------------------------------------------------
__global__ void pcl_gather(const float* __restrict__ X,
                           const int* __restrict__ idx,
                           _Float16* __restrict__ cf) {
    const int n = blockIdx.x;        // 0..1023
    const int v = n >> 6;
    const int a = n & 63;
    const int src = idx[a * NV + v];
    const float*    sp = X + (size_t)src * DD;
    _Float16*       dp = cf + (size_t)n * DD;
    dp[threadIdx.x] = (_Float16)sp[threadIdx.x];   // 256 threads == DD
}

// ---------------------------------------------------------------------------
// Kernel 3: Gram matrix logits_raw = cf * cf^T  (1024x1024, K=256) via
// V_WMMA_F32_16X16X32_F16.  One wave per 16x16 output tile; 8 waves / block.
// Operand fragments of A (16x32) and B (32x16) use the same contiguous order
// k = 16*(lane/16) + s; all per-lane loads are contiguous and 16B-aligned.
// ---------------------------------------------------------------------------
__global__ void pcl_gemm_wmma(const _Float16* __restrict__ cf,
                              float* __restrict__ logits) {
    const int wave = threadIdx.x >> 5;
    const int lane = threadIdx.x & 31;
    const int tile = blockIdx.x * 8 + wave;   // 0..4095
    const int ti = tile >> 6;                  // row tile
    const int tj = tile & 63;                  // col tile

    const int rowA = ti * 16 + (lane & 15);
    const int rowB = tj * 16 + (lane & 15);
    const int kbA  = (lane >> 4) * 16;   // K sub-offset for A fragment
    const int kbB  = (lane >> 4) * 16;   // K sub-offset for B fragment

    const _Float16* pa = cf + (size_t)rowA * DD + kbA;
    const _Float16* pb = cf + (size_t)rowB * DD + kbB;

    v8f acc = {};
#pragma unroll
    for (int kk = 0; kk < DD; kk += 32) {
        if (kk + 32 < DD) {
            __builtin_prefetch(pa + kk + 32, 0, 1);
            __builtin_prefetch(pb + kk + 32, 0, 1);
        }
        const v8h lo = *(const v8h*)(pa + kk);
        const v8h hi = *(const v8h*)(pa + kk + 8);
        const v16h Am = __builtin_shufflevector(
            lo, hi, 0, 1, 2, 3, 4, 5, 6, 7, 8, 9, 10, 11, 12, 13, 14, 15);
        const v16h Bm = *(const v16h*)(pb + kk);
        acc = __builtin_amdgcn_wmma_f32_16x16x32_f16(
            false, Am, false, Bm, (short)0, acc, false, false);
    }

    // D layout: VGPR r, lanes 0-15 -> M=r, lanes 16-31 -> M=8+r; N = lane&15
    const int col   = tj * 16 + (lane & 15);
    const int rbase = ti * 16 + (lane >> 4) * 8;
#pragma unroll
    for (int r = 0; r < 8; ++r)
        logits[(size_t)(rbase + r) * NN + col] = acc[r];
}

// ---------------------------------------------------------------------------
// Kernel 4: per-row contrastive math.  One 256-thread block per row.
// logits stored raw (dot products); scale by 1/TEMP here.
// ---------------------------------------------------------------------------
__global__ void pcl_row(const float* __restrict__ logits,
                        const int* __restrict__ valid,
                        float* __restrict__ row_loss) {
    const int i   = blockIdx.x;           // 0..1023
    const int tid = threadIdx.x;
    const int ai  = i & 63;
    const int li  = ai & 7;
    const int vi  = valid[ai];
    const float invT = 1.0f / TEMP_F;
    const float* row = logits + (size_t)i * NN;

    __shared__ float red[256];
    __shared__ int   vlab[AA];            // cached valid flags
    if (tid < AA) vlab[tid] = valid[tid];
    __syncthreads();

    // pass 1: row max of scaled logits (over ALL columns, as in reference)
    float m = -3.0e38f;
    for (int j = tid; j < NN; j += 256) m = fmaxf(m, row[j] * invT);
    red[tid] = m;
    __syncthreads();
    for (int s = 128; s > 0; s >>= 1) {
        if (tid < s) red[tid] = fmaxf(red[tid], red[tid + s]);
        __syncthreads();
    }
    m = red[0];
    __syncthreads();

    // pass 2: neg_sum = sum_j exp(l'_ij) * (1-pos_ij) * vcol_j
    float ns = 0.0f;
    for (int j = tid; j < NN; j += 256) {
        const int aj = j & 63;
        const int vj = vlab[aj];
        const float pos = (li == (aj & 7) && vi && vj) ? 1.0f : 0.0f;
        const float negm = (1.0f - pos) * (vj ? 1.0f : 0.0f);
        ns += expf(row[j] * invT - m) * negm;
    }
    red[tid] = ns;
    __syncthreads();
    for (int s = 128; s > 0; s >>= 1) {
        if (tid < s) red[tid] += red[tid + s];
        __syncthreads();
    }
    ns = red[0];
    __syncthreads();

    // pass 3: sum(mask * log_prob), sum(mask)
    float sp = 0.0f, sm = 0.0f;
    for (int j = tid; j < NN; j += 256) {
        const int aj = j & 63;
        const int vj = vlab[aj];
        const float pos = (li == (aj & 7) && vi && vj) ? 1.0f : 0.0f;
        const float maskv = (i == j) ? 0.0f : pos;
        const float lp = row[j] * invT - m;
        sp += maskv * (lp - logf(expf(lp) + ns));
        sm += maskv;
    }
    red[tid] = sp;
    __syncthreads();
    for (int s = 128; s > 0; s >>= 1) {
        if (tid < s) red[tid] += red[tid + s];
        __syncthreads();
    }
    sp = red[0];
    __syncthreads();
    red[tid] = sm;
    __syncthreads();
    for (int s = 128; s > 0; s >>= 1) {
        if (tid < s) red[tid] += red[tid + s];
        __syncthreads();
    }
    if (tid == 0) {
        const float mlpp = sp / fmaxf(red[0], 1.0f);
        const float loss = -(TEMP_F / BASE_TEMP_F) * mlpp;
        row_loss[i] = loss * (vi ? 1.0f : 0.0f);
    }
}

// ---------------------------------------------------------------------------
// Kernel 5: final deterministic scalar reduce.
// ---------------------------------------------------------------------------
__global__ void pcl_final(const float* __restrict__ row_loss,
                          const int* __restrict__ valid,
                          float* __restrict__ out) {
    __shared__ float red[256];
    const int tid = threadIdx.x;
    float s = 0.0f;
    for (int i = tid; i < NN; i += 256) s += row_loss[i];
    red[tid] = s;
    __syncthreads();
    for (int st = 128; st > 0; st >>= 1) {
        if (tid < st) red[tid] += red[tid + st];
        __syncthreads();
    }
    if (tid == 0) {
        int vs = 0;
        for (int a = 0; a < AA; ++a) vs += valid[a];
        const float denom = fmaxf((float)(vs * NV), 1.0f);
        out[0] = red[0] / denom;
    }
}

// ---------------------------------------------------------------------------
extern "C" void kernel_launch(void* const* d_in, const int* in_sizes, int n_in,
                              void* d_out, int out_size, void* d_ws, size_t ws_size,
                              hipStream_t stream) {
    (void)in_sizes; (void)n_in; (void)out_size; (void)ws_size;
    const float* X     = (const float*)d_in[0];
    const int*   y_hat = (const int*)d_in[1];
    const int*   y     = (const int*)d_in[2];
    float*       out   = (float*)d_out;

    char* w = (char*)d_ws;
    int*      idx      = (int*)w;                                  //  4 KB
    int*      valid    = idx + AA * NV;                            //  256 B
    _Float16* cf       = (_Float16*)(w + 8192);                    //  512 KB
    float*    logits   = (float*)(w + 8192 + (size_t)NN * DD * 2); //  4 MB
    float*    row_loss = logits + (size_t)NN * NN;                 //  4 KB

    pcl_select    <<<AA,      256, 0, stream>>>(y_hat, y, idx, valid);
    pcl_gather    <<<NN,      256, 0, stream>>>(X, idx, cf);
    pcl_gemm_wmma <<<NN * NN / 256 / 8 * 8 / 8, 256, 0, stream>>>(cf, logits); // 512 blocks
    pcl_row       <<<NN,      256, 0, stream>>>(logits, valid, row_loss);
    pcl_final     <<<1,       256, 0, stream>>>(row_loss, valid, out);
}
